// LSTMAttention_1297080123471
// MI455X (gfx1250) — compile-verified, matrix operand order checked by reading, not run on hardware
//
#include <hip/hip_runtime.h>
#include <hip/hip_bf16.h>

// ---------------- problem config ----------------
static constexpr int kB = 64, kT = 512, kD = 128, kH = 256, kO = 128, kSteps = 60;
static constexpr int k4H = 4 * kH;  // 1024
static constexpr int k2H = 2 * kH;  // 512

typedef __attribute__((ext_vector_type(16))) __bf16 bf16x16;
typedef __attribute__((ext_vector_type(8)))  __bf16 bf16x8;
typedef __attribute__((ext_vector_type(8)))  float  f32x8;

__device__ __forceinline__ float sigm(float x) { return 1.0f / (1.0f + __expf(-x)); }

// Opaque offset launder: defeats LICM so large loop-invariant weight streams are
// re-read from L2 each timestep instead of being hoisted and spilled to scratch.
// Laundering the *offset* (not the pointer) preserves the addrspace(1) provenance
// of the base kernel argument -> loads lower as global_load_b128 (LOADcnt only),
// not flat_load_b128 (which would also tie up DScnt).
__device__ __forceinline__ size_t opaque_off(size_t v) {
  asm volatile("" : "+v"(v));
  return v;
}

// Load one 16x32 bf16 WMMA operand fragment (A: rows=M, or B: rows=N of W[N][K]).
// Per ISA layout: lanes 0-15 hold K {k..k+7, k+16..k+23}, lanes 16-31 hold
// K {k+8..k+15, k+24..k+31}; each lane reads two contiguous 16B chunks of a row.
__device__ __forceinline__ bf16x16 frag_ld(const __bf16* rowbase, int ld, int k,
                                           int l16, int half) {
  const __bf16* p = rowbase + (size_t)l16 * ld + k + half * 8;
  bf16x8 lo = *(const bf16x8*)(p);
  bf16x8 hi = *(const bf16x8*)(p + 16);
  return __builtin_shufflevector(lo, hi, 0, 1, 2, 3, 4, 5, 6, 7,
                                 8, 9, 10, 11, 12, 13, 14, 15);
}

__device__ __forceinline__ f32x8 wmma_bf16(bf16x16 a, bf16x16 b, f32x8 c) {
  return __builtin_amdgcn_wmma_f32_16x16x32_bf16(false, a, false, b, (short)0, c,
                                                 false, false);
}

// ---------------- elementwise helpers ----------------
__global__ void cast_bf16_kernel(const float* __restrict__ s, __bf16* __restrict__ d,
                                 int n) {
  int i = blockIdx.x * blockDim.x + threadIdx.x;
  if (i < n) d[i] = (__bf16)s[i];
}

__global__ void combine_bias_kernel(const float* __restrict__ a,
                                    const float* __restrict__ b,
                                    float* __restrict__ o, int n) {
  int i = blockIdx.x * blockDim.x + threadIdx.x;
  if (i < n) o[i] = a[i] + b[i];
}

// ---------------- generic WMMA GEMM:  Out[M,N] = A[M,K](bf16) @ W[N,K]^T + bias ----
// block = 256 threads (8 waves); wave computes a 16x64 strip (A frag reused x4).
// grid = (N/64, M/128)
template <int OUT_BF16>
__global__ __launch_bounds__(256) void wmma_gemm(const __bf16* __restrict__ A,
                                                 const __bf16* __restrict__ W,
                                                 const float* __restrict__ bias,
                                                 void* __restrict__ Out, int N, int K) {
  const int wave = threadIdx.x >> 5;
  const int lane = threadIdx.x & 31;
  const int l16 = lane & 15, half = lane >> 4;
  const int m0 = blockIdx.y * 128 + wave * 16;
  const int n0 = blockIdx.x * 64;
  f32x8 acc[4] = {};
  const __bf16* arow = A + (size_t)m0 * K;
  for (int k = 0; k < K; k += 32) {
    bf16x16 af = frag_ld(arow, K, k, l16, half);
#pragma unroll
    for (int nt = 0; nt < 4; ++nt) {
      bf16x16 bfr = frag_ld(W + (size_t)(n0 + nt * 16) * K, K, k, l16, half);
      acc[nt] = wmma_bf16(af, bfr, acc[nt]);
    }
  }
#pragma unroll
  for (int nt = 0; nt < 4; ++nt) {
    const int n = n0 + nt * 16 + l16;
    const float bv = bias ? bias[n] : 0.0f;
#pragma unroll
    for (int r = 0; r < 8; ++r) {
      const int m = m0 + r + half * 8;  // C/D layout: lanes16-31 hold M+8
      const float v = acc[nt][r] + bv;
      if (OUT_BF16)
        ((__bf16*)Out)[(size_t)m * N + n] = (__bf16)v;
      else
        ((float*)Out)[(size_t)m * N + n] = v;
    }
  }
}

// ---------------- persistent bidirectional LSTM scan ----------------
// grid.x = 2 (dir 0 fwd / dir 1 bwd), block = 1024 (32 waves).
// Gates precomputed (x@Wih^T + bias) in Gin[dir][B*T][1024] bf16.
// Per step: gates += h @ Whh^T via WMMA (h bf16 in LDS); c fp32 in registers.
// Wave w owns rows [mt*16,mt*16+16) and H-columns [ng*32,ng*32+32) for ALL four
// gates (tiles at n0 = q*256 + ng*32 + j*16) -> gate math fully in registers.
// Whh is re-streamed from L2 every step (512KB, L2-resident) via opaque_off.
__global__ __launch_bounds__(1024) void lstm_scan(const __bf16* __restrict__ Gin,
                                                  size_t ginDirStride,
                                                  const __bf16* __restrict__ Whh,
                                                  size_t whhDirStride,
                                                  __bf16* __restrict__ Out, int T) {
  const int dir = blockIdx.x;
  Gin += (size_t)dir * ginDirStride;
  Whh += (size_t)dir * whhDirStride;
  const int colOff = dir * kH;
  __shared__ __bf16 hbuf[kB][kH + 8];  // pad 8 bf16 -> conflict-free frag loads
  const int tid = threadIdx.x;
  const int lane = tid & 31, l16 = lane & 15, half = lane >> 4;
  const int wv = tid >> 5, mt = wv & 3, ng = wv >> 2;
  for (int i = tid; i < kB * (kH + 8); i += 1024) (&hbuf[0][0])[i] = (__bf16)0.0f;
  f32x8 cst[2] = {};
  __syncthreads();
  for (int tt = 0; tt < T; ++tt) {
    const int t = dir ? (T - 1 - tt) : tt;
    // defeat LICM: weight fragments must be re-loaded (from L2), not spilled
    const __bf16* Wt = Whh + opaque_off((size_t)(ng * 32) * kH);
    // prefetch this step's precomputed-gate rows into cache (global_prefetch_b8)
    __builtin_prefetch((const void*)(Gin + ((size_t)(mt * 16) * T + t) * k4H), 0, 1);
    f32x8 acc[8] = {};
    for (int k = 0; k < kH; k += 32) {
      bf16x16 af = frag_ld(&hbuf[mt * 16][0], kH + 8, k, l16, half);
#pragma unroll
      for (int q = 0; q < 4; ++q)
#pragma unroll
        for (int j = 0; j < 2; ++j) {
          // row n0 = q*256 + ng*32 + j*16 ; Wt already offset by ng*32 rows
          bf16x16 bfr = frag_ld(Wt + (size_t)(q * kH + j * 16) * kH, kH, k, l16, half);
          acc[q * 2 + j] = wmma_bf16(af, bfr, acc[q * 2 + j]);
        }
    }
    __syncthreads();  // all waves done reading h of this step
#pragma unroll
    for (int j = 0; j < 2; ++j)
#pragma unroll
      for (int r = 0; r < 8; ++r) {
        const int bi = mt * 16 + r + half * 8;
        const int nc = ng * 32 + j * 16 + l16;
        const size_t grow = ((size_t)bi * T + t) * (size_t)k4H;
        const float gi = acc[0 + j][r] + (float)Gin[grow + 0 * kH + nc];
        const float gf = acc[2 + j][r] + (float)Gin[grow + 1 * kH + nc];
        const float gg = acc[4 + j][r] + (float)Gin[grow + 2 * kH + nc];
        const float go = acc[6 + j][r] + (float)Gin[grow + 3 * kH + nc];
        const float c = sigm(gf) * cst[j][r] + sigm(gi) * tanhf(gg);
        const float h = sigm(go) * tanhf(c);
        cst[j][r] = c;
        hbuf[bi][nc] = (__bf16)h;
        Out[((size_t)bi * T + t) * k2H + colOff + nc] = (__bf16)h;
      }
    __syncthreads();
  }
}

// ---------------- 60-step attention decoder ----------------
// grid.x = 2 (independent fwd/bwd scans), block = 1024.
// Per step: scores = h.keys (VALU, keys L2-resident), softmax over T,
// ctx = a @ enc (VALU), then LSTM cell gates = ctx@Wih^T + h@Whh^T via WMMA.
__global__ __launch_bounds__(1024) void attn_decoder(
    const __bf16* __restrict__ enc, const __bf16* __restrict__ keys,
    const __bf16* __restrict__ WihF, const __bf16* __restrict__ WhhF,
    const float* __restrict__ biasF, const __bf16* __restrict__ WihB,
    const __bf16* __restrict__ WhhB, const float* __restrict__ biasB,
    __bf16* __restrict__ d1) {
  const int dir = blockIdx.x;
  const __bf16* Wih = dir ? WihB : WihF;
  const __bf16* Whh = dir ? WhhB : WhhF;
  const float* bias = dir ? biasB : biasF;
  extern __shared__ char smem[];
  float*  sc   = (float*)smem;                                      // [64][516]
  __bf16* ctx  = (__bf16*)(smem + 64 * 516 * 4);                    // [64][520]
  __bf16* hb   = (__bf16*)(smem + 64 * 516 * 4 + 64 * 520 * 2);     // [64][264]
  float*  red  = (float*)(smem + 64 * 516 * 4 + 64 * 520 * 2 + 64 * 264 * 2);
  float*  red2 = red + 64 * 16;                                     // [64]
  const int tid = threadIdx.x;
  const int lane = tid & 31, l16 = lane & 15, half = lane >> 4;
  const int wv = tid >> 5, mt = wv & 3, ng = wv >> 2;
  const int b = tid >> 4, s16 = tid & 15;  // 16 threads per batch row
  for (int i = tid; i < 64 * 264; i += 1024) hb[i] = (__bf16)0.0f;
  f32x8 cst[2] = {};
  __syncthreads();
  for (int step = 0; step < kSteps; ++step) {
    // ---- scores s[b,t] = h[b] . keys[b,t] ----
    for (int tq = 0; tq < 32; ++tq) {
      const int t = s16 + tq * 16;
      const __bf16* kp = keys + ((size_t)b * kT + t) * kH;
      const __bf16* hp = hb + b * 264;
      float s = 0.0f;
      for (int kk = 0; kk < kH; kk += 8) {
        bf16x8 kv = *(const bf16x8*)(kp + kk);
        bf16x8 hv = *(const bf16x8*)(hp + kk);
#pragma unroll
        for (int u = 0; u < 8; ++u) s += (float)kv[u] * (float)hv[u];
      }
      sc[b * 516 + t] = s;
    }
    __syncthreads();
    // ---- softmax over t (per b, 16-thread cooperative) ----
    float mx = -3.0e38f;
    for (int tq = 0; tq < 32; ++tq) mx = fmaxf(mx, sc[b * 516 + s16 + tq * 16]);
    red[b * 16 + s16] = mx;
    __syncthreads();
    if (s16 == 0) {
      float m = red[b * 16];
      for (int u = 1; u < 16; ++u) m = fmaxf(m, red[b * 16 + u]);
      red2[b] = m;
    }
    __syncthreads();
    mx = red2[b];
    float sum = 0.0f;
    for (int tq = 0; tq < 32; ++tq) {
      const int idx = b * 516 + s16 + tq * 16;
      const float e = __expf(sc[idx] - mx);
      sc[idx] = e;
      sum += e;
    }
    red[b * 16 + s16] = sum;
    __syncthreads();
    if (s16 == 0) {
      float s0 = 0.0f;
      for (int u = 0; u < 16; ++u) s0 += red[b * 16 + u];
      red2[b] = 1.0f / s0;
    }
    __syncthreads();
    const float inv = red2[b];
    for (int tq = 0; tq < 32; ++tq) sc[b * 516 + s16 + tq * 16] *= inv;
    __syncthreads();
    // ---- context ctx[b,e] = sum_t a[b,t] * enc[b,t,e] (enc L2-resident) ----
    float cacc[32];
#pragma unroll
    for (int u = 0; u < 32; ++u) cacc[u] = 0.0f;
    for (int t = 0; t < kT; ++t) {
      const float a = sc[b * 516 + t];
      const __bf16* ep = enc + ((size_t)b * kT + t) * k2H + s16;
#pragma unroll
      for (int eq = 0; eq < 32; ++eq) cacc[eq] += a * (float)ep[eq * 16];
    }
#pragma unroll
    for (int eq = 0; eq < 32; ++eq) ctx[b * 520 + s16 + eq * 16] = (__bf16)cacc[eq];
    __syncthreads();
    // ---- LSTM cell via WMMA: gates = ctx@Wih^T + h@Whh^T + bias ----
    // defeat LICM on the weight streams (L2-resident, re-read per step)
    const __bf16* WihT = Wih + opaque_off((size_t)(ng * 32) * k2H);
    const __bf16* WhhT = Whh + opaque_off((size_t)(ng * 32) * kH);
    f32x8 acc[8] = {};
    for (int k = 0; k < k2H; k += 32) {
      bf16x16 af = frag_ld(ctx + (size_t)(mt * 16) * 520, 520, k, l16, half);
#pragma unroll
      for (int q = 0; q < 4; ++q)
#pragma unroll
        for (int j = 0; j < 2; ++j) {
          bf16x16 bfr =
              frag_ld(WihT + (size_t)(q * kH + j * 16) * k2H, k2H, k, l16, half);
          acc[q * 2 + j] = wmma_bf16(af, bfr, acc[q * 2 + j]);
        }
    }
    for (int k = 0; k < kH; k += 32) {
      bf16x16 af = frag_ld(hb + (size_t)(mt * 16) * 264, 264, k, l16, half);
#pragma unroll
      for (int q = 0; q < 4; ++q)
#pragma unroll
        for (int j = 0; j < 2; ++j) {
          bf16x16 bfr =
              frag_ld(WhhT + (size_t)(q * kH + j * 16) * kH, kH, k, l16, half);
          acc[q * 2 + j] = wmma_bf16(af, bfr, acc[q * 2 + j]);
        }
    }
    __syncthreads();  // everyone finished reading hb/ctx
#pragma unroll
    for (int j = 0; j < 2; ++j)
#pragma unroll
      for (int r = 0; r < 8; ++r) {
        const int bi = mt * 16 + r + half * 8;
        const int nc = ng * 32 + j * 16 + l16;
        const float gi = acc[0 + j][r] + bias[0 * kH + nc];
        const float gf = acc[2 + j][r] + bias[1 * kH + nc];
        const float gg = acc[4 + j][r] + bias[2 * kH + nc];
        const float go = acc[6 + j][r] + bias[3 * kH + nc];
        const float c = sigm(gf) * cst[j][r] + sigm(gi) * tanhf(gg);
        const float h = sigm(go) * tanhf(c);
        cst[j][r] = c;
        hb[bi * 264 + nc] = (__bf16)h;
        const int tpos = dir ? (kSteps - 1 - step) : step;  // bwd written reversed
        d1[((size_t)bi * kSteps + tpos) * k2H + dir * kH + nc] = (__bf16)h;
      }
    __syncthreads();
  }
}

// ---------------- host orchestration ----------------
extern "C" void kernel_launch(void* const* d_in, const int* in_sizes, int n_in,
                              void* d_out, int out_size, void* d_ws, size_t ws_size,
                              hipStream_t stream) {
  (void)in_sizes; (void)n_in; (void)out_size; (void)ws_size;
  const float* x_f      = (const float*)d_in[0];
  const float* enc0_Wih = (const float*)d_in[1];
  const float* enc0_Whh = (const float*)d_in[2];
  const float* enc0_b   = (const float*)d_in[3];
  const float* encL_Wih = (const float*)d_in[4];
  const float* encL_Whh = (const float*)d_in[5];
  const float* encL_b   = (const float*)d_in[6];
  const float* attn_W   = (const float*)d_in[7];
  const float* attn_b   = (const float*)d_in[8];
  const float* d1f_Wih  = (const float*)d_in[9];
  const float* d1f_Whh  = (const float*)d_in[10];
  const float* d1f_b    = (const float*)d_in[11];
  const float* d1b_Wih  = (const float*)d_in[12];
  const float* d1b_Whh  = (const float*)d_in[13];
  const float* d1b_b    = (const float*)d_in[14];
  const float* dec2_Wih = (const float*)d_in[15];
  const float* dec2_Whh = (const float*)d_in[16];
  const float* dec2_b   = (const float*)d_in[17];
  const float* lin_W    = (const float*)d_in[18];
  const float* lin_b    = (const float*)d_in[19];

  // --- workspace bump allocator (deterministic layout every call) ---
  char* p = (char*)d_ws;
  auto take = [&](size_t bytes) -> void* {
    void* r = (void*)p;
    p += (bytes + 255) & ~(size_t)255;
    return r;
  };
  const size_t BT = (size_t)kB * kT;      // 32768
  const size_t BS = (size_t)kB * kSteps;  // 3840
  __bf16* xbf       = (__bf16*)take(BT * kD * 2);
  __bf16* encA      = (__bf16*)take(BT * k2H * 2);
  __bf16* encB      = (__bf16*)take(BT * k2H * 2);
  __bf16* Gin       = (__bf16*)take(2 * BT * k4H * 2);  // both directions
  __bf16* keysb     = (__bf16*)take(BT * kH * 2);
  __bf16* d1        = (__bf16*)take(BS * k2H * 2);
  __bf16* d2A       = (__bf16*)take(BS * k2H * 2);
  __bf16* d2B       = (__bf16*)take(BS * k2H * 2);
  __bf16* Gd        = (__bf16*)take(2 * BS * k4H * 2);
  __bf16* w_e0_Wih  = (__bf16*)take((size_t)2 * 1024 * 128 * 2);
  __bf16* w_e0_Whh  = (__bf16*)take((size_t)2 * 1024 * 256 * 2);
  __bf16* w_eL_Wih  = (__bf16*)take((size_t)4 * 1024 * 512 * 2);
  __bf16* w_eL_Whh  = (__bf16*)take((size_t)4 * 1024 * 256 * 2);
  __bf16* w_attn    = (__bf16*)take((size_t)256 * 512 * 2);
  __bf16* w_d1f_Wih = (__bf16*)take((size_t)1024 * 512 * 2);
  __bf16* w_d1f_Whh = (__bf16*)take((size_t)1024 * 256 * 2);
  __bf16* w_d1b_Wih = (__bf16*)take((size_t)1024 * 512 * 2);
  __bf16* w_d1b_Whh = (__bf16*)take((size_t)1024 * 256 * 2);
  __bf16* w_d2_Wih  = (__bf16*)take((size_t)4 * 1024 * 512 * 2);
  __bf16* w_d2_Whh  = (__bf16*)take((size_t)4 * 1024 * 256 * 2);
  __bf16* w_lin     = (__bf16*)take((size_t)128 * 512 * 2);
  float* cb_e0  = (float*)take(2 * 1024 * 4);
  float* cb_eL  = (float*)take(4 * 1024 * 4);
  float* cb_d1f = (float*)take(1024 * 4);
  float* cb_d1b = (float*)take(1024 * 4);
  float* cb_d2  = (float*)take(4 * 1024 * 4);

  auto cast = [&](const float* s, __bf16* d, int n) {
    cast_bf16_kernel<<<dim3((n + 255) / 256), dim3(256), 0, stream>>>(s, d, n);
  };
  auto comb = [&](const float* pair, float* dst) {  // dst = pair[0:1024] + pair[1024:2048]
    combine_bias_kernel<<<dim3(4), dim3(256), 0, stream>>>(pair, pair + 1024, dst, 1024);
  };

  // --- fp32 -> bf16 weight/activation staging ---
  cast(x_f, xbf, (int)(BT * kD));
  cast(enc0_Wih, w_e0_Wih, 2 * 1024 * 128);
  cast(enc0_Whh, w_e0_Whh, 2 * 1024 * 256);
  cast(encL_Wih, w_eL_Wih, 4 * 1024 * 512);
  cast(encL_Whh, w_eL_Whh, 4 * 1024 * 256);
  cast(attn_W, w_attn, 256 * 512);
  cast(d1f_Wih, w_d1f_Wih, 1024 * 512);
  cast(d1f_Whh, w_d1f_Whh, 1024 * 256);
  cast(d1b_Wih, w_d1b_Wih, 1024 * 512);
  cast(d1b_Whh, w_d1b_Whh, 1024 * 256);
  cast(dec2_Wih, w_d2_Wih, 4 * 1024 * 512);
  cast(dec2_Whh, w_d2_Whh, 4 * 1024 * 256);
  cast(lin_W, w_lin, 128 * 512);
  // combined biases b_ih + b_hh
  comb(enc0_b + 0, cb_e0 + 0);
  comb(enc0_b + 2048, cb_e0 + 1024);
  for (int l = 0; l < 2; ++l)
    for (int d = 0; d < 2; ++d)
      comb(encL_b + (size_t)((l * 2 + d) * 2) * 1024, cb_eL + (l * 2 + d) * 1024);
  comb(d1f_b, cb_d1f);
  comb(d1b_b, cb_d1b);
  for (int l = 0; l < 2; ++l)
    for (int d = 0; d < 2; ++d)
      comb(dec2_b + (size_t)((l * 2 + d) * 2) * 1024, cb_d2 + (l * 2 + d) * 1024);

  const dim3 blk256(256);
  const size_t ginStride = BT * (size_t)k4H;
  const size_t gdStride  = BS * (size_t)k4H;

  // --- encoder layer 0 (input D=128) ---
  for (int d = 0; d < 2; ++d)
    wmma_gemm<1><<<dim3(k4H / 64, (int)(BT / 128)), blk256, 0, stream>>>(
        xbf, w_e0_Wih + (size_t)d * 1024 * 128, cb_e0 + d * 1024,
        Gin + (size_t)d * ginStride, k4H, kD);
  lstm_scan<<<dim3(2), dim3(1024), 0, stream>>>(Gin, ginStride, w_e0_Whh,
                                                (size_t)1024 * 256, encA, kT);
  // --- encoder layers 1..2 (input 2H=512) ---
  __bf16* encIn = encA;
  __bf16* encOut = encB;
  for (int l = 0; l < 2; ++l) {
    for (int d = 0; d < 2; ++d)
      wmma_gemm<1><<<dim3(k4H / 64, (int)(BT / 128)), blk256, 0, stream>>>(
          encIn, w_eL_Wih + (size_t)(l * 2 + d) * 1024 * 512,
          cb_eL + (l * 2 + d) * 1024, Gin + (size_t)d * ginStride, k4H, k2H);
    lstm_scan<<<dim3(2), dim3(1024), 0, stream>>>(
        Gin, ginStride, w_eL_Whh + (size_t)l * 2 * 1024 * 256, (size_t)1024 * 256,
        encOut, kT);
    __bf16* tmp = encIn; encIn = encOut; encOut = tmp;
  }
  // final enc is in encIn

  // --- attention keys = enc @ attn_W^T + attn_b ---
  wmma_gemm<1><<<dim3(kH / 64, (int)(BT / 128)), blk256, 0, stream>>>(
      encIn, w_attn, attn_b, keysb, kH, k2H);

  // --- 60-step attention decoder (fwd + bwd, 2 persistent workgroups) ---
  const int decSmem = 64 * 516 * 4 + 64 * 520 * 2 + 64 * 264 * 2 + 64 * 16 * 4 + 64 * 4;
  (void)hipFuncSetAttribute((const void*)attn_decoder,
                            hipFuncAttributeMaxDynamicSharedMemorySize, decSmem);
  attn_decoder<<<dim3(2), dim3(1024), decSmem, stream>>>(
      encIn, keysb, w_d1f_Wih, w_d1f_Whh, cb_d1f, w_d1b_Wih, w_d1b_Whh, cb_d1b, d1);

  // --- dec2: 2-layer bidirectional LSTM over 60 steps ---
  __bf16* din = d1;
  __bf16* douts[2] = {d2A, d2B};
  for (int l = 0; l < 2; ++l) {
    for (int d = 0; d < 2; ++d)
      wmma_gemm<1><<<dim3(k4H / 64, (int)(BS / 128)), blk256, 0, stream>>>(
          din, w_d2_Wih + (size_t)(l * 2 + d) * 1024 * 512,
          cb_d2 + (l * 2 + d) * 1024, Gd + (size_t)d * gdStride, k4H, k2H);
    lstm_scan<<<dim3(2), dim3(1024), 0, stream>>>(
        Gd, gdStride, w_d2_Whh + (size_t)l * 2 * 1024 * 256, (size_t)1024 * 256,
        douts[l], kSteps);
    din = douts[l];
  }

  // --- linear head -> fp32 output [B,60,O] ---
  wmma_gemm<0><<<dim3(kO / 64, (int)(BS / 128)), blk256, 0, stream>>>(
      din, w_lin, lin_b, d_out, kO, k2H);
}
